// AlgebraicAttention_48756468744650
// MI455X (gfx1250) — matrix operand order repaired and checked by reading, not compiled, not run on hardware
//
#include <hip/hip_runtime.h>
#include <hip/hip_bf16.h>

// ---------------------------------------------------------------------------
// CDNA5 / gfx1250 wave32 WMMA implementation of ALiBi attention with
// rational (algebraic-sigmoid) softmax.  B=2, T=2048, C=1024, H=16, D=64
// Stage 1: qkv = x @ w_qkv.T  (bf16 WMMA, double-buffered LDS) -> Q,K,Vt bf16
// Stage 2: flash-style attention, K/V chunks streamed by the Tensor Data
//          Mover (tensor_load_to_lds + s_wait_tensorcnt, double-buffered)
// Stage 3: out = attn @ w_out.T -> f32
// ---------------------------------------------------------------------------

typedef __attribute__((ext_vector_type(16))) __bf16          v16bf;
typedef __attribute__((ext_vector_type(16))) unsigned short  v16u;
typedef __attribute__((ext_vector_type(8)))  float           v8f;
typedef __attribute__((ext_vector_type(4)))  unsigned int    v4u;
typedef __attribute__((ext_vector_type(8)))  int             v8i;
typedef __attribute__((ext_vector_type(4)))  int             v4i;

#define T_SEQ 2048
#define D_HEAD 64
#define N_HEAD 16
#define C_DIM 1024

#if defined(__has_builtin)
#if __has_builtin(__builtin_amdgcn_tensor_load_to_lds)
#define HAVE_TDM 1
#endif
#endif
#ifndef HAVE_TDM
#define HAVE_TDM 0
#endif

// round-to-nearest-even f32 -> bf16
__device__ __forceinline__ unsigned short f2bf(float f) {
  union { float f; unsigned int u; } c; c.f = f;
  unsigned int u = c.u;
  return (unsigned short)((u + 0x7FFFu + ((u >> 16) & 1u)) >> 16);
}
__device__ __forceinline__ unsigned int pk2(float a, float b) {
  return (unsigned int)f2bf(a) | ((unsigned int)f2bf(b) << 16);
}

__device__ __forceinline__ v8f wmma_bf16(v16u a, v16u b, v8f c) {
  return __builtin_amdgcn_wmma_f32_16x16x32_bf16(
      false, __builtin_bit_cast(v16bf, a),
      false, __builtin_bit_cast(v16bf, b),
      (short)0, c, false, false);
}

// load 32B fragment (16 bf16) from two 16B-aligned LDS locations
__device__ __forceinline__ v16u ldfrag(const unsigned short* p0,
                                       const unsigned short* p1) {
  union { uint4 q[2]; v16u v; } u;
  u.q[0] = *(const uint4*)p0;
  u.q[1] = *(const uint4*)p1;
  return u.v;
}

// convert 16 f32 -> 16 bf16 and store 32B to LDS
__device__ __forceinline__ void cvt_store(unsigned short* dst, const float4* f) {
  uint4 p0, p1;
  p0.x = pk2(f[0].x, f[0].y); p0.y = pk2(f[0].z, f[0].w);
  p0.z = pk2(f[1].x, f[1].y); p0.w = pk2(f[1].z, f[1].w);
  p1.x = pk2(f[2].x, f[2].y); p1.y = pk2(f[2].z, f[2].w);
  p1.z = pk2(f[3].x, f[3].y); p1.w = pk2(f[3].z, f[3].w);
  ((uint4*)dst)[0] = p0; ((uint4*)dst)[1] = p1;
}

#if HAVE_TDM
// low 32 bits of a flat shared-aperture pointer == logical LDS byte address
__device__ __forceinline__ unsigned lds_lo32(const void* p) {
  return (unsigned)(uintptr_t)p;
}
// Tensor Data Mover: 2-D tile of 2-byte elements, global -> LDS.
// D# per CDNA5 ISA 8.3/8.4: group0 = {count|flags, lds_addr, global_addr,
// type=2}; group1 = {data_size=2B, tensor dims, tile dims, dim0 stride}.
// This toolchain exposes the 6-arg builtin: (g0, g1, g2, g3, extra, cpol);
// groups 2/3 and the extra operand are zero (tile_dim2..4 unused).
__device__ __forceinline__ void tdm_load_2d(unsigned lds_addr, const void* g,
                                            unsigned tile_d0, unsigned tile_d1,
                                            unsigned long long stride0,
                                            unsigned tens_d0, unsigned tens_d1) {
  unsigned long long ga = (unsigned long long)(uintptr_t)g;
  v4u g0;
  g0[0] = 1u;                                           // count=1, user D#
  g0[1] = lds_addr;                                     // LDS dest (bytes)
  g0[2] = (unsigned)(ga & 0xFFFFFFFFu);                 // global_addr[31:0]
  g0[3] = (unsigned)((ga >> 32) & 0x01FFFFFFu) | (2u << 30);  // [56:32]|type=2
  v8i g1;
  g1[0] = (int)(1u << 16);                              // data_size=1 -> 2B
  g1[1] = (int)((tens_d0 & 0xFFFFu) << 16);             // tensor_dim0 lo16
  g1[2] = (int)((tens_d0 >> 16) | ((tens_d1 & 0xFFFFu) << 16));
  g1[3] = (int)((tens_d1 >> 16) | (tile_d0 << 16));     // tile_dim0
  g1[4] = (int)tile_d1;                                 // tile_dim1 (dim2=0)
  g1[5] = (int)(stride0 & 0xFFFFFFFFull);               // dim0 stride lo32
  g1[6] = (int)((stride0 >> 32) & 0xFFFFull);           // dim0 stride hi16
  g1[7] = 0;
  v4i z4 = {0, 0, 0, 0};
  v8i z8i = {0, 0, 0, 0, 0, 0, 0, 0};
  __builtin_amdgcn_tensor_load_to_lds(g0, g1, z4, z4, z8i, 0);
}
#endif

// ---------------------------------------------------------------------------
// Generic Y[m,n] = sum_k A[m,k] * B[n,k] GEMM, 128x128 block tile, 8 waves
// (2x4), wave tile 64x32 (4x2 wmma accums), K=32 per stage, double-buffered
// LDS.  EPI 0: scatter bf16 into Q/K/Vt.  EPI 1: f32 store.
// ---------------------------------------------------------------------------
template <int AF32, int BF32, int EPI>
__global__ __launch_bounds__(256) void gemm_kernel(
    const void* __restrict__ Ap, const void* __restrict__ Bp,
    float* __restrict__ Yf,
    unsigned short* __restrict__ Qo, unsigned short* __restrict__ Ko,
    unsigned short* __restrict__ Vto,
    int M, int N, int K) {
  __shared__ __align__(16) unsigned short As[2][128 * 32];
  __shared__ __align__(16) unsigned short Bs[2][128 * 32];

  const int tid  = threadIdx.x;
  const int lane = tid & 31;
  const int wave = tid >> 5;
  const int lm   = lane & 15;
  const int lhi  = lane >> 4;
  const int wm0  = (wave >> 2) * 64;
  const int wn0  = (wave & 3) * 32;
  const int m0   = blockIdx.y * 128;
  const int n0   = blockIdx.x * 128;

  const int lrow  = tid >> 1;  // 0..127
  const int lpart = tid & 1;   // 0..1 (16 halves each)
  const size_t aBase = (size_t)(m0 + lrow) * K + lpart * 16;
  const size_t bBase = (size_t)(n0 + lrow) * K + lpart * 16;
  const int lofs = lrow * 32 + lpart * 16;

  float4 fa[4], fb[4];
  uint4  qa[2], qb[2];

  // prologue: fetch k0=0 tile and stage into buffer 0
  if (AF32) {
    const float4* s = (const float4*)((const float*)Ap + aBase);
    fa[0] = s[0]; fa[1] = s[1]; fa[2] = s[2]; fa[3] = s[3];
  } else {
    const uint4* s = (const uint4*)((const unsigned short*)Ap + aBase);
    qa[0] = s[0]; qa[1] = s[1];
  }
  if (BF32) {
    const float4* s = (const float4*)((const float*)Bp + bBase);
    fb[0] = s[0]; fb[1] = s[1]; fb[2] = s[2]; fb[3] = s[3];
  } else {
    const uint4* s = (const uint4*)((const unsigned short*)Bp + bBase);
    qb[0] = s[0]; qb[1] = s[1];
  }
  if (AF32) cvt_store(&As[0][lofs], fa);
  else { ((uint4*)&As[0][lofs])[0] = qa[0]; ((uint4*)&As[0][lofs])[1] = qa[1]; }
  if (BF32) cvt_store(&Bs[0][lofs], fb);
  else { ((uint4*)&Bs[0][lofs])[0] = qb[0]; ((uint4*)&Bs[0][lofs])[1] = qb[1]; }

  v8f acc[4][2];
  const v8f z8 = {0.f, 0.f, 0.f, 0.f, 0.f, 0.f, 0.f, 0.f};
#pragma unroll
  for (int i = 0; i < 4; ++i)
#pragma unroll
    for (int j = 0; j < 2; ++j) acc[i][j] = z8;

  int buf = 0;
  for (int k0 = 0; k0 < K; k0 += 32) {
    __syncthreads();
    const bool hasNext = (k0 + 32 < K);
    if (hasNext) {  // start next tile's global fetch (overlaps with WMMA)
      if (AF32) {
        const float4* s = (const float4*)((const float*)Ap + aBase + k0 + 32);
        fa[0] = s[0]; fa[1] = s[1]; fa[2] = s[2]; fa[3] = s[3];
      } else {
        const uint4* s =
            (const uint4*)((const unsigned short*)Ap + aBase + k0 + 32);
        qa[0] = s[0]; qa[1] = s[1];
      }
      if (BF32) {
        const float4* s = (const float4*)((const float*)Bp + bBase + k0 + 32);
        fb[0] = s[0]; fb[1] = s[1]; fb[2] = s[2]; fb[3] = s[3];
      } else {
        const uint4* s =
            (const uint4*)((const unsigned short*)Bp + bBase + k0 + 32);
        qb[0] = s[0]; qb[1] = s[1];
      }
    }

    v16u afr[4], bfr[2];
#pragma unroll
    for (int i = 0; i < 4; ++i) {
      const unsigned short* ar = &As[buf][(wm0 + i * 16 + lm) * 32];
      afr[i] = ldfrag(ar + 8 * lhi, ar + 16 + 8 * lhi);
    }
#pragma unroll
    for (int j = 0; j < 2; ++j) {
      const unsigned short* br = &Bs[buf][(wn0 + j * 16 + lm) * 32 + 8 * lhi];
      bfr[j] = ldfrag(br, br + 16);
    }
#pragma unroll
    for (int i = 0; i < 4; ++i)
#pragma unroll
      for (int j = 0; j < 2; ++j)
        acc[i][j] = wmma_bf16(afr[i], bfr[j], acc[i][j]);

    if (hasNext) {  // stage next tile into the other buffer
      if (AF32) cvt_store(&As[buf ^ 1][lofs], fa);
      else {
        ((uint4*)&As[buf ^ 1][lofs])[0] = qa[0];
        ((uint4*)&As[buf ^ 1][lofs])[1] = qa[1];
      }
      if (BF32) cvt_store(&Bs[buf ^ 1][lofs], fb);
      else {
        ((uint4*)&Bs[buf ^ 1][lofs])[0] = qb[0];
        ((uint4*)&Bs[buf ^ 1][lofs])[1] = qb[1];
      }
    }
    buf ^= 1;
  }

  // epilogue (C layout: lane l holds col (l&15), rows g (+8 for hi lanes))
#pragma unroll
  for (int i = 0; i < 4; ++i) {
#pragma unroll
    for (int j = 0; j < 2; ++j) {
#pragma unroll
      for (int g = 0; g < 8; ++g) {
        const int m = m0 + wm0 + i * 16 + g + 8 * lhi;
        const int n = n0 + wn0 + j * 16 + lm;
        const float v = acc[i][j][g];
        if (EPI == 0) {
          const int b = m >> 11, t = m & 2047;
          const int s = n >> 10, h = (n >> 6) & 15, d = n & 63;
          const unsigned short bv = f2bf(v);
          if (s == 0)
            Qo[((size_t)((b * N_HEAD + h) * T_SEQ + t)) * D_HEAD + d] = bv;
          else if (s == 1)
            Ko[((size_t)((b * N_HEAD + h) * T_SEQ + t)) * D_HEAD + d] = bv;
          else
            Vto[((size_t)((b * N_HEAD + h) * D_HEAD + d)) * T_SEQ + t] = bv;
        } else {
          Yf[(size_t)m * N + n] = v;
        }
      }
    }
  }
}

// ---------------------------------------------------------------------------
// Attention: one block = 128 query rows of one (b,h); each wave owns 16 rows.
// K/V chunks of 64 keys are streamed into double-buffered LDS by the Tensor
// Data Mover (wave 0 issues; pipelined with s_wait_tensorcnt).  Rational
// softmax is bounded -> no running-max rescaling; full-T loop so masked
// columns' num(-10000) contributes to the normalizer like the reference.
// ---------------------------------------------------------------------------
__global__ __launch_bounds__(256) void attn_kernel(
    const unsigned short* __restrict__ Q, const unsigned short* __restrict__ Km,
    const unsigned short* __restrict__ Vt, const float* __restrict__ slopes,
    unsigned short* __restrict__ attn) {
  __shared__ __align__(16) unsigned short Ks[2][64 * 64];  // [key][d]
  __shared__ __align__(16) unsigned short Vs[2][64 * 64];  // [d][key]
  __shared__ __align__(16) unsigned short Ps[8 * 1024];    // per-wave probs

  const int tid  = threadIdx.x;
  const int lane = tid & 31;
  const int wave = tid >> 5;
  const int lm   = lane & 15;
  const int lhi  = lane >> 4;
  const int b    = blockIdx.z;
  const int h    = blockIdx.y;
  const int r0   = blockIdx.x * 128 + wave * 16;

  const unsigned short* Qp = Q  + ((size_t)(b * N_HEAD + h) * T_SEQ) * D_HEAD;
  const unsigned short* Kp = Km + ((size_t)(b * N_HEAD + h) * T_SEQ) * D_HEAD;
  const unsigned short* Vp = Vt + ((size_t)(b * N_HEAD + h) * D_HEAD) * T_SEQ;
  const float slope = slopes[h];
  const float scale = 0.125f;  // 1/sqrt(64)

  // resident Q fragments: 16 rows x 64 d = two 16x32 A frags
  v16u aq[2];
  {
    const unsigned short* qrow = Qp + (size_t)(r0 + lm) * D_HEAD;
#pragma unroll
    for (int ks = 0; ks < 2; ++ks)
      aq[ks] = ldfrag(qrow + ks * 32 + 8 * lhi, qrow + ks * 32 + 16 + 8 * lhi);
  }

  float rs[8];
#pragma unroll
  for (int g = 0; g < 8; ++g) rs[g] = 0.f;
  v8f o[4];
  const v8f z8 = {0.f, 0.f, 0.f, 0.f, 0.f, 0.f, 0.f, 0.f};
#pragma unroll
  for (int d = 0; d < 4; ++d) o[d] = z8;

  unsigned short* pw = &Ps[wave * 1024];
#if !HAVE_TDM
  const int ldrow  = tid >> 2;  // 0..63
  const int ldpart = tid & 3;   // 0..3
#endif

#if HAVE_TDM
  if (wave == 0) {  // prefetch chunk 0 (TDM ignores EXEC; uniform branch)
    tdm_load_2d(lds_lo32(&Ks[0][0]), Kp, 64, 64, 64ull, 64, 64);
    tdm_load_2d(lds_lo32(&Vs[0][0]), Vp, 64, 64, (unsigned long long)T_SEQ,
                64, 64);
  }
#endif
  int buf = 0;
  for (int s0 = 0; s0 < T_SEQ; s0 += 64) {
#if HAVE_TDM
    if (wave == 0) {
      if (s0 + 64 < T_SEQ) {  // stream chunk i+1 while consuming chunk i
        tdm_load_2d(lds_lo32(&Ks[buf ^ 1][0]),
                    Kp + (size_t)(s0 + 64) * D_HEAD, 64, 64, 64ull, 64, 64);
        tdm_load_2d(lds_lo32(&Vs[buf ^ 1][0]), Vp + (s0 + 64), 64, 64,
                    (unsigned long long)T_SEQ, 64, 64);
        __builtin_amdgcn_s_wait_tensorcnt(2);  // chunk i landed
      } else {
        __builtin_amdgcn_s_wait_tensorcnt(0);
      }
    }
    __syncthreads();
#else
    __syncthreads();
    {
      const uint4* ksrc =
          (const uint4*)(Kp + (size_t)(s0 + ldrow) * D_HEAD + ldpart * 16);
      uint4* kdst = (uint4*)(&Ks[buf][ldrow * 64 + ldpart * 16]);
      kdst[0] = ksrc[0]; kdst[1] = ksrc[1];
      const uint4* vsrc =
          (const uint4*)(Vp + (size_t)ldrow * T_SEQ + s0 + ldpart * 16);
      uint4* vdst = (uint4*)(&Vs[buf][ldrow * 64 + ldpart * 16]);
      vdst[0] = vsrc[0]; vdst[1] = vsrc[1];
    }
    __syncthreads();
#endif

    // ---- scores -> num -> per-wave LDS scratch (A-fragment layout) ----
#pragma unroll
    for (int ns = 0; ns < 4; ++ns) {
      const unsigned short* kr = &Ks[buf][(ns * 16 + lm) * 64];
      v16u bk0 = ldfrag(kr + 8 * lhi, kr + 16 + 8 * lhi);
      v16u bk1 = ldfrag(kr + 32 + 8 * lhi, kr + 48 + 8 * lhi);
      v8f sc = z8;
      sc = wmma_bf16(aq[0], bk0, sc);
      sc = wmma_bf16(aq[1], bk1, sc);

      const int sg    = s0 + ns * 16 + lm;         // global key index
      const int laneA = 16 * (lm >> 3);            // A-frag lane-half select
      const int ib    = (lm & 7) + 8 * (ns & 1);   // A-frag half index
#pragma unroll
      for (int g = 0; g < 8; ++g) {
        const int mg = r0 + g + 8 * lhi;           // global query index
        float x = sc[g] * scale - slope * fabsf((float)(mg - sg));
        if (sg > mg) x = -10000.0f;                // causal mask (pre-num)
        const float num = 0.5f * (1.0f + x * rsqrtf(1.0f + x * x));
        rs[g] += num;
        pw[(ns >> 1) * 512 + (g + 8 * lhi + laneA) * 16 + ib] = f2bf(num);
      }
    }
    // same-wave LDS store->load dependency (DS ops in-order per wave)
    asm volatile("s_wait_dscnt 0" ::: "memory");

    v16u ap[2];
#pragma unroll
    for (int blk = 0; blk < 2; ++blk) {
      const unsigned short* pr = pw + blk * 512 + lane * 16;
      ap[blk] = ldfrag(pr, pr + 8);
    }
    // ---- out += num @ V ----
#pragma unroll
    for (int ds = 0; ds < 4; ++ds) {
      const unsigned short* vr = &Vs[buf][(ds * 16 + lm) * 64];
      v16u bv0 = ldfrag(vr + 8 * lhi, vr + 16 + 8 * lhi);
      v16u bv1 = ldfrag(vr + 32 + 8 * lhi, vr + 48 + 8 * lhi);
      o[ds] = wmma_bf16(ap[0], bv0, o[ds]);
      o[ds] = wmma_bf16(ap[1], bv1, o[ds]);
    }
    __syncthreads();  // all waves done with buf before TDM refills it
    buf ^= 1;
  }

  // row-sum reduction across each 16-lane half (xor masks stay in-half and
  // broadcast the result to every lane of the half)
#pragma unroll
  for (int g = 0; g < 8; ++g) {
    float v = rs[g];
    v += __shfl_xor(v, 1);
    v += __shfl_xor(v, 2);
    v += __shfl_xor(v, 4);
    v += __shfl_xor(v, 8);
    rs[g] = 1.0f / (v + 1e-6f);
  }

  // write attn[b*T + t][h*64 + d] as bf16
#pragma unroll
  for (int ds = 0; ds < 4; ++ds) {
#pragma unroll
    for (int g = 0; g < 8; ++g) {
      const int t = r0 + g + 8 * lhi;
      const int c = h * D_HEAD + ds * 16 + lm;
      attn[(size_t)(b * T_SEQ + t) * C_DIM + c] = f2bf(o[ds][g] * rs[g]);
    }
  }
}

// ---------------------------------------------------------------------------
extern "C" void kernel_launch(void* const* d_in, const int* in_sizes, int n_in,
                              void* d_out, int out_size, void* d_ws,
                              size_t ws_size, hipStream_t stream) {
  const float* x     = (const float*)d_in[0];  // [2,2048,1024]
  const float* w_qkv = (const float*)d_in[1];  // [3072,1024]
  const float* w_out = (const float*)d_in[2];  // [1024,1024]
  const float* slope = (const float*)d_in[3];  // [16]
  float* out = (float*)d_out;                  // [2,2048,1024]

  const size_t QKV_ELEMS = (size_t)2 * N_HEAD * T_SEQ * D_HEAD;  // 4M
  unsigned short* Q    = (unsigned short*)d_ws;
  unsigned short* Kk   = Q + QKV_ELEMS;
  unsigned short* Vt   = Kk + QKV_ELEMS;
  unsigned short* attn = Vt + QKV_ELEMS;

  dim3 g1(3072 / 128, 4096 / 128);
  gemm_kernel<1, 1, 0><<<g1, 256, 0, stream>>>(x, w_qkv, nullptr, Q, Kk, Vt,
                                               4096, 3072, 1024);
  dim3 g2(T_SEQ / 128, N_HEAD, 2);
  attn_kernel<<<g2, 256, 0, stream>>>(Q, Kk, Vt, slope, attn);
  dim3 g3(1024 / 128, 4096 / 128);
  gemm_kernel<0, 1, 1><<<g3, 256, 0, stream>>>(attn, w_out, out, nullptr,
                                               nullptr, nullptr, 4096, 1024,
                                               1024);
}
